// NGCF_matrix_12575664242933
// MI455X (gfx1250) — matrix operand order repaired and checked by reading, not compiled, not run on hardware
//
#include <hip/hip_runtime.h>
#include <hip/hip_bf16.h>

#define N_NODES 50000
#define DIM     64
#define N_EDGES 1250000
#define N_LAYER 3
#define BATCH   4096

typedef __attribute__((ext_vector_type(16))) __bf16 v16bf;
typedef __attribute__((ext_vector_type(8)))  float  v8f;

// native f32->bf16 (RNE); lets the compiler use gfx1250 hardware cvt
__device__ __forceinline__ __bf16 f2bf(float f) { return (__bf16)f; }

// ---------------- zero fill ----------------
__global__ void zero_f32(float* __restrict__ p, long long n) {
    long long i = (long long)blockIdx.x * blockDim.x + threadIdx.x;
    if (i < n) p[i] = 0.0f;
}

// ---------------- SpMM: agg[rows[e]] += vals[e] * x[cols[e]] ----------------
// one thread per (edge, dim); vals/rows/cols loads broadcast across the 64
// consecutive lanes of an edge; gather + atomic scatter are 256B/edge bursts
// that stay resident in the 192MB L2 (whole emb table = 12.8MB).
__global__ __launch_bounds__(256) void spmm_atomic(
    const float* __restrict__ vals, const int* __restrict__ rows,
    const int* __restrict__ cols, const float* __restrict__ x,
    float* __restrict__ agg, long long total) {
    long long gid = (long long)blockIdx.x * blockDim.x + threadIdx.x;
    if (gid >= total) return;
    int e = (int)(gid >> 6);
    int d = (int)(gid & 63);
    float v = vals[e];
    int r = rows[e], c = cols[e];
    atomicAdd(&agg[(long long)r * DIM + d], v * x[(long long)c * DIM + d]);
}

// ---------------- fused dense layer via WMMA ----------------
// nxt = leaky_relu( (agg+cur) @ W1^T + b1  +  agg * (cur @ W2^T + b2) )
// one wave per 16-node tile; K=64 -> 2 x wmma_f32_16x16x32_bf16 per 16x16
// output tile, 4 column tiles, two GEMMs -> 16 WMMAs per wave.
__global__ __launch_bounds__(32) void dense_layer_wmma(
    const float* __restrict__ cur, const float* __restrict__ agg,
    const float* __restrict__ w1, const float* __restrict__ b1,
    const float* __restrict__ w2, const float* __restrict__ b2,
    float* __restrict__ nxt) {
    const int row_base = blockIdx.x * 16;
    const int lane = threadIdx.x;       // 0..31
    const int r    = lane & 15;         // A row within tile / B,C column
    const int hb   = lane >> 4;         // lane-half

    const long long node = row_base + r;
    const float* curp = cur + node * DIM;
    const float* aggp = agg + node * DIM;

    // A operands: a_s = (agg+cur) tile, a_e = cur tile, 16x64 as 2 K-chunks.
    // A 16x32 k order chosen here (same as B): elem i -> K = c*32 + hb*16 + i  (16 contig)
    // => per (c, half) an 8-float contiguous run: base = c*32 + hb*16 + h2*8
    v16bf a_s[2] = {}, a_e[2] = {};
#pragma unroll
    for (int c = 0; c < 2; ++c) {
#pragma unroll
        for (int h2 = 0; h2 < 2; ++h2) {
            const int base = c * 32 + hb * 16 + h2 * 8;
            float eb[8], gb[8];
            *(float4*)&eb[0] = *(const float4*)(curp + base);
            *(float4*)&eb[4] = *(const float4*)(curp + base + 4);
            *(float4*)&gb[0] = *(const float4*)(aggp + base);
            *(float4*)&gb[4] = *(const float4*)(aggp + base + 4);
#pragma unroll
            for (int i = 0; i < 8; ++i) {
                a_e[c][h2 * 8 + i] = f2bf(eb[i]);
                a_s[c][h2 * 8 + i] = f2bf(gb[i] + eb[i]);
            }
        }
    }

#pragma unroll
    for (int tn = 0; tn < 4; ++tn) {
        const int ncol = tn * 16 + r;   // this lane's output / B column
        // B 32x16 k order chosen here (same as A): elem i -> K = c*32 + hb*16 + i  (16 contig)
        // Y = X @ W^T  =>  B[k][n] = W[n][k]
        v16bf bw1[2] = {}, bw2[2] = {};
#pragma unroll
        for (int c = 0; c < 2; ++c) {
            const int kb = c * 32 + hb * 16;
            float wb1[16], wb2[16];
            *(float4*)&wb1[0]  = *(const float4*)(w1 + ncol * DIM + kb);
            *(float4*)&wb1[4]  = *(const float4*)(w1 + ncol * DIM + kb + 4);
            *(float4*)&wb1[8]  = *(const float4*)(w1 + ncol * DIM + kb + 8);
            *(float4*)&wb1[12] = *(const float4*)(w1 + ncol * DIM + kb + 12);
            *(float4*)&wb2[0]  = *(const float4*)(w2 + ncol * DIM + kb);
            *(float4*)&wb2[4]  = *(const float4*)(w2 + ncol * DIM + kb + 4);
            *(float4*)&wb2[8]  = *(const float4*)(w2 + ncol * DIM + kb + 8);
            *(float4*)&wb2[12] = *(const float4*)(w2 + ncol * DIM + kb + 12);
#pragma unroll
            for (int i = 0; i < 16; ++i) {
                bw1[c][i] = f2bf(wb1[i]);
                bw2[c][i] = f2bf(wb2[i]);
            }
        }

        // preload epilogue operands so they overlap the WMMAs and issue as
        // one load clause instead of 8 serialized round-trips
        const float bias1 = b1[ncol];
        const float bias2 = b2[ncol];
        float av[8];
#pragma unroll
        for (int j = 0; j < 8; ++j)
            av[j] = agg[(long long)(row_base + hb * 8 + j) * DIM + ncol];

        v8f acc1 = {}, acc2 = {};
        acc1 = __builtin_amdgcn_wmma_f32_16x16x32_bf16(false, a_s[0], false, bw1[0],
                                                       (short)0, acc1, false, false);
        acc1 = __builtin_amdgcn_wmma_f32_16x16x32_bf16(false, a_s[1], false, bw1[1],
                                                       (short)0, acc1, false, false);
        acc2 = __builtin_amdgcn_wmma_f32_16x16x32_bf16(false, a_e[0], false, bw2[0],
                                                       (short)0, acc2, false, false);
        acc2 = __builtin_amdgcn_wmma_f32_16x16x32_bf16(false, a_e[1], false, bw2[1],
                                                       (short)0, acc2, false, false);

        // C/D layout: VGPR j -> row m = hb*8 + j, col = ncol
        float outv[8];
#pragma unroll
        for (int j = 0; j < 8; ++j) {
            float sx = acc1[j] + bias1;
            float tt = acc2[j] + bias2;
            float y  = sx + av[j] * tt;
            outv[j] = (y >= 0.0f) ? y : 0.01f * y;
        }
#pragma unroll
        for (int j = 0; j < 8; ++j)
            nxt[(long long)(row_base + hb * 8 + j) * DIM + ncol] = outv[j];
    }
}

// ---------------- per-stage logit accumulation ----------------
// concat-dot == sum over stages of per-stage dots; one wave per batch elem.
__global__ __launch_bounds__(128) void logits_accum(
    const float* __restrict__ x, const int* __restrict__ user,
    const int* __restrict__ pos, const int* __restrict__ neg,
    float* __restrict__ posl, float* __restrict__ negl) {
    int wid  = (int)((blockIdx.x * (long long)blockDim.x + threadIdx.x) >> 5);
    int lane = threadIdx.x & 31;
    if (wid >= BATCH) return;
    long long u = user[wid], p = pos[wid], ng = neg[wid];
    float dp = 0.0f, dn = 0.0f;
#pragma unroll
    for (int i = lane; i < DIM; i += 32) {
        float uv = x[u * DIM + i];
        dp += uv * x[p * DIM + i];
        dn += uv * x[ng * DIM + i];
    }
#pragma unroll
    for (int off = 16; off > 0; off >>= 1) {
        dp += __shfl_down(dp, off, 32);
        dn += __shfl_down(dn, off, 32);
    }
    if (lane == 0) { posl[wid] += dp; negl[wid] += dn; }
}

// ---------------- loss reduction ----------------
__global__ __launch_bounds__(256) void loss_reduce(
    const float* __restrict__ posl, const float* __restrict__ negl,
    float* __restrict__ out) {
    __shared__ float sm[256];
    float acc = 0.0f;
    for (int i = threadIdx.x; i < BATCH; i += 256) {
        float z = negl[i] - posl[i];                 // -(pos - neg)
        // -log_sigmoid(diff) = softplus(-diff), stable
        acc += fmaxf(z, 0.0f) + log1pf(expf(-fabsf(z)));
    }
    sm[threadIdx.x] = acc;
    __syncthreads();
    for (int s = 128; s > 0; s >>= 1) {
        if (threadIdx.x < s) sm[threadIdx.x] += sm[threadIdx.x + s];
        __syncthreads();
    }
    if (threadIdx.x == 0) out[0] = sm[0];
}

extern "C" void kernel_launch(void* const* d_in, const int* in_sizes, int n_in,
                              void* d_out, int out_size, void* d_ws, size_t ws_size,
                              hipStream_t stream) {
    (void)in_sizes; (void)n_in; (void)out_size; (void)ws_size;
    const float* emb  = (const float*)d_in[0];
    const float* w1_w = (const float*)d_in[1];
    const float* w1_b = (const float*)d_in[2];
    const float* w2_w = (const float*)d_in[3];
    const float* w2_b = (const float*)d_in[4];
    const float* vals = (const float*)d_in[5];
    const int*   rows = (const int*)d_in[6];
    const int*   cols = (const int*)d_in[7];
    const int*   user = (const int*)d_in[8];
    const int*   pos  = (const int*)d_in[9];
    const int*   neg  = (const int*)d_in[10];

    const long long ND = (long long)N_NODES * DIM;
    float* cur  = (float*)d_ws;
    float* nxt  = cur + ND;
    float* agg  = nxt + ND;
    float* posl = agg + ND;
    float* negl = posl + BATCH;

    // cur = emb (don't mutate inputs)
    hipMemcpyAsync(cur, emb, ND * sizeof(float), hipMemcpyDeviceToDevice, stream);

    // zero logit accumulators (posl/negl contiguous)
    zero_f32<<<(2 * BATCH + 255) / 256, 256, 0, stream>>>(posl, 2 * BATCH);

    // stage 0 logits
    logits_accum<<<BATCH / 4, 128, 0, stream>>>(cur, user, pos, neg, posl, negl);

    const long long spmm_total = (long long)N_EDGES * DIM;
    const int spmm_blocks = (int)((spmm_total + 255) / 256);

    for (int l = 0; l < N_LAYER; ++l) {
        zero_f32<<<(int)((ND + 255) / 256), 256, 0, stream>>>(agg, ND);
        spmm_atomic<<<spmm_blocks, 256, 0, stream>>>(vals, rows, cols, cur, agg, spmm_total);
        dense_layer_wmma<<<N_NODES / 16, 32, 0, stream>>>(
            cur, agg,
            w1_w + (long long)l * DIM * DIM, w1_b + (long long)l * DIM,
            w2_w + (long long)l * DIM * DIM, w2_b + (long long)l * DIM,
            nxt);
        logits_accum<<<BATCH / 4, 128, 0, stream>>>(nxt, user, pos, neg, posl, negl);
        float* t = cur; cur = nxt; nxt = t;
    }

    loss_reduce<<<1, 256, 0, stream>>>(posl, negl, (float*)d_out);
}
